// VectorQuantizer_79285096284401
// MI455X (gfx1250) — matrix operand order repaired and checked by reading, not compiled, not run on hardware
//
#include <hip/hip_runtime.h>
#include <math.h>

typedef __attribute__((ext_vector_type(16))) _Float16 v16h;
typedef __attribute__((ext_vector_type(8)))  _Float16 v8h;
typedef __attribute__((ext_vector_type(8)))  float    v8f;

#define NUM_EMB   1024
#define EMB_DIM   64
#define LDS_STRIDE 72                      // padded row stride in halves (144 B)
#define N_ROWS    (64 * 4096)
#define ROWS_PER_WAVE   16
#define WAVES_PER_BLOCK 8
#define ROWS_PER_BLOCK  (ROWS_PER_WAVE * WAVES_PER_BLOCK)   // 128
#define NUM_BLOCKS      (N_ROWS / ROWS_PER_BLOCK)           // 2048
#define CODE_TILES      (NUM_EMB / 16)                      // 64

#define SMEM_EMB_BYTES  (NUM_EMB * LDS_STRIDE * 2)          // 147456
#define SMEM_BYTES      (SMEM_EMB_BYTES + NUM_EMB * 4)      // + e2[] = 151552

// ---------------------------------------------------------------------------
// Kernel 0: zero histogram + loss accumulator (must happen every call).
// ---------------------------------------------------------------------------
__global__ __launch_bounds__(256)
void vq_init(float* __restrict__ counts, float* __restrict__ lossAcc) {
    int k = blockIdx.x * blockDim.x + threadIdx.x;
    if (k < NUM_EMB) {
        counts[k] = 0.f;
        if (k == 0) *lossAcc = 0.f;
    }
}

// ---------------------------------------------------------------------------
// Kernel 1: WMMA distance GEMM + argmin + gather + partial reductions.
// Codebook staged once per block into LDS as f16 (padded stride), with
// ||e_k||^2 computed during the fill.  One wave handles 16 rows of z.
// Distance surrogate: ||e||^2 - 2*z.e  (||z||^2 is row-constant).
// ---------------------------------------------------------------------------
__global__ __launch_bounds__(256)
void vq_main(const float* __restrict__ z,
             const float* __restrict__ emb,
             float* __restrict__ zq,
             float* __restrict__ counts,
             float* __restrict__ lossAcc) {
    extern __shared__ char smem[];
    _Float16* embh = (_Float16*)smem;                       // [1024][72] f16
    float*    e2s  = (float*)(smem + SMEM_EMB_BYTES);       // [1024] f32

    const int tid  = threadIdx.x;
    const int lane = tid & 31;
    const int wave = tid >> 5;
    const int row0 = blockIdx.x * ROWS_PER_BLOCK + wave * ROWS_PER_WAVE;

    // ---- cooperative codebook stage: f32 -> f16 into LDS, + ||e||^2 -------
    for (int r = tid; r < NUM_EMB; r += 256) {
        const float* er  = emb + (size_t)r * EMB_DIM;
        _Float16*    dst = embh + (size_t)r * LDS_STRIDE;
        float ssum = 0.f;
        #pragma unroll
        for (int d = 0; d < EMB_DIM; d += 8) {
            float4 a = *(const float4*)(er + d);
            float4 b = *(const float4*)(er + d + 4);
            v8h h;
            h[0] = (_Float16)a.x; h[1] = (_Float16)a.y;
            h[2] = (_Float16)a.z; h[3] = (_Float16)a.w;
            h[4] = (_Float16)b.x; h[5] = (_Float16)b.y;
            h[6] = (_Float16)b.z; h[7] = (_Float16)b.w;
            ssum += a.x*a.x + a.y*a.y + a.z*a.z + a.w*a.w
                  + b.x*b.x + b.y*b.y + b.z*b.z + b.w*b.w;
            *(v8h*)(dst + d) = h;
        }
        e2s[r] = ssum;
    }
    __syncthreads();

    // ---- A fragments: 16 z rows, f32 -> f16, per 16-bit A 16x32 layout ----
    // lanes 0-15: M=lane, K in {0..7, 16..23}; lanes 16-31: M=lane-16, K+8.
    const int arow = lane & 15;
    const int akb  = (lane >> 4) << 3;          // 0 or 8
    v16h afrag[2];
    #pragma unroll
    for (int s = 0; s < 2; ++s) {
        const float* zr = z + (size_t)(row0 + arow) * EMB_DIM + s * 32 + akb;
        float buf[16];
        *(float4*)(buf + 0)  = *(const float4*)(zr + 0);
        *(float4*)(buf + 4)  = *(const float4*)(zr + 4);
        *(float4*)(buf + 8)  = *(const float4*)(zr + 16);
        *(float4*)(buf + 12) = *(const float4*)(zr + 20);
        #pragma unroll
        for (int j = 0; j < 16; ++j) afrag[s][j] = (_Float16)buf[j];
    }

    // ---- running per-row argmin state (rows striped over 8 acc VGPRs) ----
    float bestD[8];
    int   bestI[8];
    #pragma unroll
    for (int r = 0; r < 8; ++r) { bestD[r] = 3.4e38f; bestI[r] = 0; }

    // B layout: lane&15 = code column; lanes 0-15 hold K in {0..7, 16..23}; lanes 16-31 K+8.
    const int bcol = lane & 15;
    const int bkb  = (lane >> 4) << 3;          // 0 or 8

    for (int t = 0; t < CODE_TILES; ++t) {
        const int code = t * 16 + bcol;
        const _Float16* bp = embh + (size_t)code * LDS_STRIDE + bkb;

        v8h lo0 = *(const v8h*)(bp);
        v8h hi0 = *(const v8h*)(bp + 16);
        v8h lo1 = *(const v8h*)(bp + 32);
        v8h hi1 = *(const v8h*)(bp + 48);
        v16h b0 = __builtin_shufflevector(lo0, hi0, 0,1,2,3,4,5,6,7,
                                                    8,9,10,11,12,13,14,15);
        v16h b1 = __builtin_shufflevector(lo1, hi1, 0,1,2,3,4,5,6,7,
                                                    8,9,10,11,12,13,14,15);
        const float ee = e2s[code];

        v8f acc = {};
        acc = __builtin_amdgcn_wmma_f32_16x16x32_f16(
                  false, afrag[0], false, b0, (short)0, acc, false, false);
        acc = __builtin_amdgcn_wmma_f32_16x16x32_f16(
                  false, afrag[1], false, b1, (short)0, acc, false, false);

        #pragma unroll
        for (int r = 0; r < 8; ++r) {
            float d = ee - 2.0f * acc[r];
            if (d < bestD[r]) { bestD[r] = d; bestI[r] = code; }
        }
    }

    // ---- min-reduce across the 16 lanes sharing each row (tie: low index) --
    #pragma unroll
    for (int r = 0; r < 8; ++r) {
        #pragma unroll
        for (int m = 1; m < 16; m <<= 1) {
            float od = __shfl_xor(bestD[r], m, 32);
            int   oi = __shfl_xor(bestI[r], m, 32);
            if (od < bestD[r] || (od == bestD[r] && oi < bestI[r])) {
                bestD[r] = od; bestI[r] = oi;
            }
        }
    }

    // ---- gather winning codebook rows, emit z_q, accumulate loss/hist -----
    float lsum = 0.f;
    for (int m = 0; m < 16; ++m) {
        const int r   = m & 7;
        const int src = (m < 8) ? 0 : 16;   // rows 0-7 live in lanes 0-15 (r=M)
        const int idx = __shfl(bestI[r], src, 32);

        const float2 ev = ((const float2*)(emb + (size_t)idx * EMB_DIM))[lane];
        const float2 zv = ((const float2*)(z + (size_t)(row0 + m) * EMB_DIM))[lane];
        ((float2*)(zq + (size_t)(row0 + m) * EMB_DIM))[lane] = ev;

        const float dx = ev.x - zv.x, dy = ev.y - zv.y;
        lsum += dx * dx + dy * dy;

        if (lane == 0) atomicAdd(&counts[idx], 1.0f);
    }

    #pragma unroll
    for (int m = 1; m < 32; m <<= 1) lsum += __shfl_xor(lsum, m, 32);
    if (lane == 0) atomicAdd(lossAcc, lsum);
}

// ---------------------------------------------------------------------------
// Kernel 2: scalars -> vq_loss, perplexity (appended after z_q in d_out).
// ---------------------------------------------------------------------------
__global__ __launch_bounds__(256)
void vq_finalize(const float* __restrict__ counts,
                 const float* __restrict__ lossAcc,
                 float* __restrict__ out_scalars) {
    __shared__ float sred[256];
    const int tid = threadIdx.x;
    float s = 0.f;
    for (int k = tid; k < NUM_EMB; k += 256) {
        float avg = counts[k] * (1.0f / (float)N_ROWS);
        s += avg * logf(avg + 1e-5f);
    }
    sred[tid] = s;
    __syncthreads();
    for (int off = 128; off > 0; off >>= 1) {
        if (tid < off) sred[tid] += sred[tid + off];
        __syncthreads();
    }
    if (tid == 0) {
        out_scalars[0] = 0.25f * (*lossAcc) *
                         (1.0f / ((float)N_ROWS * (float)EMB_DIM));
        out_scalars[1] = expf(-sred[0]);
    }
}

// ---------------------------------------------------------------------------
extern "C" void kernel_launch(void* const* d_in, const int* in_sizes, int n_in,
                              void* d_out, int out_size, void* d_ws, size_t ws_size,
                              hipStream_t stream) {
    const float* z   = (const float*)d_in[0];   // [64*4096, 64] f32
    const float* emb = (const float*)d_in[1];   // [1024, 64]   f32

    float* counts  = (float*)d_ws;              // 1024 f32
    float* lossAcc = counts + NUM_EMB;          // 1  f32

    float* zq          = (float*)d_out;                      // N_ROWS*EMB_DIM
    float* out_scalars = zq + (size_t)N_ROWS * EMB_DIM;      // [vq_loss, perplexity]

    vq_init<<<(NUM_EMB + 255) / 256, 256, 0, stream>>>(counts, lossAcc);
    vq_main<<<NUM_BLOCKS, 256, SMEM_BYTES, stream>>>(z, emb, zq, counts, lossAcc);
    vq_finalize<<<1, 256, 0, stream>>>(counts, lossAcc, out_scalars);
}
